// GraphAttentionLayer_14104672600610
// MI455X (gfx1250) — matrix operand order repaired and checked by reading, not compiled, not run on hardware
//
#include <hip/hip_runtime.h>
#include <math.h>

// ---------------- CDNA5 WMMA types ----------------
typedef __attribute__((ext_vector_type(16))) __bf16 bv16;
typedef __attribute__((ext_vector_type(8)))  float  v8f;

union BV16 { bv16 v; unsigned short u[16]; };

#define D256 256
#define WT_STRIDE 264   // halfs; 528B rows -> bank = 132*e + k/2 (mod 64): conflict-free b128 frags

__device__ __forceinline__ float fast_tanh(float x) {
#if __has_builtin(__builtin_amdgcn_tanhf)
  return __builtin_amdgcn_tanhf(x);          // native v_tanh_f32 (gfx1250 TRANS op)
#else
  // branch-free: tanh(x) = sign(x) * (1 - 2/(exp(2|x|)+1))
  float ax = __builtin_fabsf(x);
  float e = __expf(2.0f * ax);
  float t = 1.0f - 2.0f / (e + 1.0f);
  return __builtin_copysignf(t, x);
#endif
}

__device__ __forceinline__ unsigned short f2bf(float f) {
  union { float f; unsigned u; } c; c.f = f;
  unsigned r = c.u + 0x7FFFu + ((c.u >> 16) & 1u);   // round-to-nearest-even
  return (unsigned short)(r >> 16);
}

__device__ __forceinline__ void load_cvt8(const float* __restrict__ p, unsigned short* dst) {
  union { float4 v; float f[4]; } a, b;
  a.v = ((const float4*)p)[0];
  b.v = ((const float4*)p)[1];
#pragma unroll
  for (int i = 0; i < 4; ++i) { dst[i] = f2bf(a.f[i]); dst[4 + i] = f2bf(b.f[i]); }
}

// A fragment for v_wmma_f32_16x16x32_bf16: lane(0-15)=row M, halves split K:
// lanes 0-15: u[0..15]=K(kb+0..15); lanes 16-31: u[0..15]=K(kb+16..31).
__device__ __forceinline__ void load_a_frag(const float* __restrict__ arow, int kbase, int half, BV16& af) {
  int k0 = kbase + half * 16;
  load_cvt8(arow + k0,      af.u);
  load_cvt8(arow + k0 + 8,  af.u + 8);
}

// B fragment from LDS-transposed matrix Bt[col][k] (bf16, row stride WT_STRIDE):
// lane(0-15)=col N, u[0..15] = K(kb+half*16 + 0..15) -> two 16B ds loads.
__device__ __forceinline__ void load_b_frag(const unsigned short* __restrict__ brow, int kc, BV16& bf) {
  const uint4* p = (const uint4*)(brow + kc * 32);
  *(uint4*)&bf.u[0] = p[0];
  *(uint4*)&bf.u[8] = p[1];
}

// =====================================================================
// K1: scores[b,n,m] = sum_e tanh( (X[b] @ (x_n .* Wp))[m,e] + bp[e] ) * w_att[e]
// one block per (b,n); 8 waves x 32-row strips; 69 GFLOP total in bf16 WMMA
// =====================================================================
__global__ void __launch_bounds__(256) k_scores(const float* __restrict__ x,
                                                const float* __restrict__ Wp,
                                                const float* __restrict__ bp,
                                                const float* __restrict__ watt,
                                                float* __restrict__ scores) {
  extern __shared__ char smem[];
  float* xs = (float*)smem;                                  // 256 f32
  unsigned short* Wt = (unsigned short*)(smem + 1024);       // [e][d] bf16, stride 264

  const int blk = blockIdx.x;            // b*256 + n
  const int b = blk >> 8, n = blk & 255;
  const int t = threadIdx.x;

  const float* xb = x + (size_t)b * D256 * D256;
  xs[t] = xb[(size_t)n * D256 + t];
  __syncthreads();

  { // stage Wt[e][d] = bf16(Wp[d,e] * x_n[d]); rotate start d to dodge bank conflicts
    const int e = t;
    unsigned short* wrow = Wt + e * WT_STRIDE;
    const int d0 = (t >> 4) * 16;
    for (int i = 0; i < 256; i += 2) {
      int d = (d0 + i) & 255;
      unsigned short h0 = f2bf(Wp[(size_t)d * D256 + e] * xs[d]);
      unsigned short h1 = f2bf(Wp[(size_t)(d + 1) * D256 + e] * xs[d + 1]);
      *(unsigned*)(wrow + d) = (unsigned)h0 | ((unsigned)h1 << 16);
    }
  }
  __syncthreads();

  const int wave = t >> 5, lane = t & 31;
  const int half = lane >> 4, l16 = lane & 15;
  const int mbase = wave * 32;

  // preload per-lane epilogue constants (e = nt*16 + l16), mt-invariant
  float bpv[16], wav[16];
#pragma unroll
  for (int nt = 0; nt < 16; ++nt) {
    bpv[nt] = bp[nt * 16 + l16];
    wav[nt] = watt[nt * 16 + l16];
  }

#pragma unroll
  for (int mt = 0; mt < 2; ++mt) {
    const float* arow = xb + (size_t)(mbase + mt * 16 + l16) * D256;
    BV16 af[8];
#pragma unroll
    for (int kc = 0; kc < 8; ++kc) load_a_frag(arow, kc * 32, half, af[kc]);

    float part[8] = {0.f, 0.f, 0.f, 0.f, 0.f, 0.f, 0.f, 0.f};
    for (int nt = 0; nt < 16; ++nt) {
      const unsigned short* brow = Wt + (nt * 16 + l16) * WT_STRIDE + half * 16;
      v8f acc = {0.f, 0.f, 0.f, 0.f, 0.f, 0.f, 0.f, 0.f};
      // software-pipelined: issue next B-frag loads before current WMMA
      BV16 bfA, bfB;
      load_b_frag(brow, 0, bfA);
#pragma unroll
      for (int kc = 0; kc < 8; ++kc) {
        BV16& cur = (kc & 1) ? bfB : bfA;
        BV16& nxt = (kc & 1) ? bfA : bfB;
        if (kc < 7) load_b_frag(brow, kc + 1, nxt);
        acc = __builtin_amdgcn_wmma_f32_16x16x32_bf16(false, af[kc].v, false, cur.v,
                                                      (short)0, acc, false, false);
      }
      const float bpe = bpv[nt], wae = wav[nt];
#pragma unroll
      for (int j = 0; j < 8; ++j) part[j] += wae * fast_tanh(acc[j] + bpe);
    }
    // reduce over the 16 N-lanes of each half-wave
#pragma unroll
    for (int j = 0; j < 8; ++j) {
      float v = part[j];
      v += __shfl_xor(v, 1, 32);
      v += __shfl_xor(v, 2, 32);
      v += __shfl_xor(v, 4, 32);
      v += __shfl_xor(v, 8, 32);
      part[j] = v;
    }
#pragma unroll
    for (int j = 0; j < 8; ++j) {
      if (l16 == j) {
        int m = mbase + mt * 16 + j + half * 8;   // C layout: VGPR j -> row j (h0) / j+8 (h1)
        scores[(size_t)blk * D256 + m] = part[j];
      }
    }
  }
}

// =====================================================================
// K2: softmax over axis n (per (b,m) column), in place
// =====================================================================
__global__ void __launch_bounds__(256) k_softmax(float* __restrict__ s) {
  const int b = blockIdx.x >> 2;
  const int m0 = (blockIdx.x & 3) * 64;
  const int t = threadIdx.x;
  const int mc = m0 + (t & 63);
  const int g = t >> 6;                 // n-quarter 0..3
  float* sb = s + (size_t)b * 65536;

  __shared__ float red[4][64];

  float vals[64];
  float mx = -3.4e38f;
#pragma unroll
  for (int i = 0; i < 64; ++i) {
    float v = sb[(size_t)(g * 64 + i) * D256 + mc];
    vals[i] = v;
    mx = fmaxf(mx, v);
  }
  red[g][t & 63] = mx;
  __syncthreads();
  float cmax = fmaxf(fmaxf(red[0][t & 63], red[1][t & 63]),
                     fmaxf(red[2][t & 63], red[3][t & 63]));
  __syncthreads();

  float sum = 0.f;
#pragma unroll
  for (int i = 0; i < 64; ++i) { vals[i] = __expf(vals[i] - cmax); sum += vals[i]; }
  red[g][t & 63] = sum;
  __syncthreads();
  float inv = 1.f / (red[0][t & 63] + red[1][t & 63] + red[2][t & 63] + red[3][t & 63]);

#pragma unroll
  for (int i = 0; i < 64; ++i) sb[(size_t)(g * 64 + i) * D256 + mc] = vals[i] * inv;
}

// =====================================================================
// K3: agg[b] = Att[b] @ X[b]   (WMMA; stage X[b]^T bf16 in LDS)
// =====================================================================
__global__ void __launch_bounds__(256) k_agg(const float* __restrict__ att,
                                             const float* __restrict__ x,
                                             float* __restrict__ agg) {
  extern __shared__ char smem[];
  unsigned short* Xt = (unsigned short*)smem;   // [d][m] bf16

  const int b = blockIdx.x >> 1;
  const int rbase0 = (blockIdx.x & 1) * 128;
  const int t = threadIdx.x;
  const float* xb = x + (size_t)b * 65536;
  const float* ab = att + (size_t)b * 65536;

  { // Xt[d][m] = bf16(X[b][m][d])
    const int d = t;
    unsigned short* row = Xt + d * WT_STRIDE;
    const int m0 = (t >> 4) * 16;
    for (int i = 0; i < 256; i += 2) {
      int m = (m0 + i) & 255;
      unsigned short h0 = f2bf(xb[(size_t)m * D256 + d]);
      unsigned short h1 = f2bf(xb[(size_t)(m + 1) * D256 + d]);
      *(unsigned*)(row + m) = (unsigned)h0 | ((unsigned)h1 << 16);
    }
  }
  __syncthreads();

  const int wave = t >> 5, lane = t & 31;
  const int half = lane >> 4, l16 = lane & 15;
  const int rbase = rbase0 + wave * 16;

  const float* arow = ab + (size_t)(rbase + l16) * D256;
  BV16 af[8];
#pragma unroll
  for (int kc = 0; kc < 8; ++kc) load_a_frag(arow, kc * 32, half, af[kc]);

  float* ob = agg + (size_t)b * 65536;
  for (int nt = 0; nt < 16; ++nt) {
    const int e = nt * 16 + l16;
    const unsigned short* brow = Xt + e * WT_STRIDE + half * 16;
    v8f acc = {0.f, 0.f, 0.f, 0.f, 0.f, 0.f, 0.f, 0.f};
    BV16 bfA, bfB;
    load_b_frag(brow, 0, bfA);
#pragma unroll
    for (int kc = 0; kc < 8; ++kc) {
      BV16& cur = (kc & 1) ? bfB : bfA;
      BV16& nxt = (kc & 1) ? bfA : bfB;
      if (kc < 7) load_b_frag(brow, kc + 1, nxt);
      acc = __builtin_amdgcn_wmma_f32_16x16x32_bf16(false, af[kc].v, false, cur.v,
                                                    (short)0, acc, false, false);
    }
#pragma unroll
    for (int j = 0; j < 8; ++j)
      ob[(size_t)(rbase + j + half * 8) * D256 + e] = acc[j];
  }
}

// =====================================================================
// K4: pre = agg@W1 + x@W2 + (b1+b2)   (fused dual-GEMM; stage W1^T,W2^T)
// single software-pipelined 16-step WMMA chain per N-tile
// =====================================================================
__global__ void __launch_bounds__(256) k_out(const float* __restrict__ agg,
                                             const float* __restrict__ x,
                                             const float* __restrict__ W1,
                                             const float* __restrict__ b1,
                                             const float* __restrict__ W2,
                                             const float* __restrict__ b2,
                                             float* __restrict__ pre) {
  extern __shared__ char smem[];
  unsigned short* W1t = (unsigned short*)smem;
  unsigned short* W2t = W1t + D256 * WT_STRIDE;

  const int t = threadIdx.x;
  { // stage both transposed weights
    const int e = t;
    unsigned short* r1 = W1t + e * WT_STRIDE;
    unsigned short* r2 = W2t + e * WT_STRIDE;
    const int d0 = (t >> 4) * 16;
    for (int i = 0; i < 256; i += 2) {
      int d = (d0 + i) & 255;
      unsigned a0 = (unsigned)f2bf(W1[(size_t)d * D256 + e]) |
                    ((unsigned)f2bf(W1[(size_t)(d + 1) * D256 + e]) << 16);
      unsigned c0 = (unsigned)f2bf(W2[(size_t)d * D256 + e]) |
                    ((unsigned)f2bf(W2[(size_t)(d + 1) * D256 + e]) << 16);
      *(unsigned*)(r1 + d) = a0;
      *(unsigned*)(r2 + d) = c0;
    }
  }
  __syncthreads();

  const int wave = t >> 5, lane = t & 31;
  const int half = lane >> 4, l16 = lane & 15;
  const int rbase = blockIdx.x * 128 + wave * 16;

  const float* garow = agg + (size_t)(rbase + l16) * D256;
  const float* xarow = x   + (size_t)(rbase + l16) * D256;
  BV16 afg[8], afx[8];
#pragma unroll
  for (int kc = 0; kc < 8; ++kc) {
    load_a_frag(garow, kc * 32, half, afg[kc]);
    load_a_frag(xarow, kc * 32, half, afx[kc]);
  }

  for (int nt = 0; nt < 16; ++nt) {
    const int e = nt * 16 + l16;
    const unsigned short* br1 = W1t + e * WT_STRIDE + half * 16;
    const unsigned short* br2 = W2t + e * WT_STRIDE + half * 16;
    v8f acc = {0.f, 0.f, 0.f, 0.f, 0.f, 0.f, 0.f, 0.f};
    BV16 bfA, bfB;
    load_b_frag(br1, 0, bfA);
#pragma unroll
    for (int s = 0; s < 16; ++s) {
      BV16& cur = (s & 1) ? bfB : bfA;
      BV16& nxt = (s & 1) ? bfA : bfB;
      if (s < 15) {
        const unsigned short* nb = (s + 1 < 8) ? br1 : br2;
        load_b_frag(nb, (s + 1) & 7, nxt);
      }
      const BV16& a = (s < 8) ? afg[s] : afx[s & 7];
      acc = __builtin_amdgcn_wmma_f32_16x16x32_bf16(false, a.v, false, cur.v,
                                                    (short)0, acc, false, false);
    }
    const float bb = b1[e] + b2[e];
#pragma unroll
    for (int j = 0; j < 8; ++j)
      pre[(size_t)(rbase + j + half * 8) * D256 + e] = acc[j] + bb;
  }
}

// =====================================================================
// K5: per-column sum / sumsq over 2048 rows (deterministic, no atomics)
// =====================================================================
__global__ void __launch_bounds__(256) k_stats(const float* __restrict__ pre,
                                               float* __restrict__ stats) {
  const int e = blockIdx.x, t = threadIdx.x;
  __shared__ float r1[256], r2[256];
  float s1 = 0.f, s2 = 0.f;
#pragma unroll
  for (int i = 0; i < 8; ++i) {
    float v = pre[(size_t)(t + i * 256) * D256 + e];
    s1 += v; s2 += v * v;
  }
  r1[t] = s1; r2[t] = s2;
  __syncthreads();
  for (int off = 128; off > 0; off >>= 1) {
    if (t < off) { r1[t] += r1[t + off]; r2[t] += r2[t + off]; }
    __syncthreads();
  }
  if (t == 0) { stats[e] = r1[0]; stats[256 + e] = r2[0]; }
}

// =====================================================================
// K6: training-mode BatchNorm + SELU -> d_out
// =====================================================================
__global__ void __launch_bounds__(256) k_bn_selu(const float* __restrict__ pre,
                                                 const float* __restrict__ stats,
                                                 const float* __restrict__ gamma,
                                                 const float* __restrict__ beta,
                                                 float* __restrict__ out) {
  const int t = threadIdx.x;
  const size_t idx = (size_t)blockIdx.x * 256 + t;
  const int e = t;                       // last dim == blockDim
  const float inv_n = 1.0f / 2048.0f;
  float mean = stats[e] * inv_n;
  float var = stats[256 + e] * inv_n - mean * mean;
  float v = (pre[idx] - mean) * rsqrtf(var + 1e-5f) * gamma[e] + beta[e];
  const float alpha = 1.6732632423543772f, scale = 1.0507009873554805f;
  out[idx] = scale * (v > 0.f ? v : alpha * (__expf(v) - 1.f));
}

// =====================================================================
extern "C" void kernel_launch(void* const* d_in, const int* in_sizes, int n_in,
                              void* d_out, int out_size, void* d_ws, size_t ws_size,
                              hipStream_t stream) {
  (void)in_sizes; (void)n_in; (void)out_size; (void)ws_size;
  const float* x     = (const float*)d_in[0];
  const float* Wp    = (const float*)d_in[1];
  const float* bp    = (const float*)d_in[2];
  const float* watt  = (const float*)d_in[3];
  const float* W1    = (const float*)d_in[4];
  const float* b1    = (const float*)d_in[5];
  const float* W2    = (const float*)d_in[6];
  const float* b2    = (const float*)d_in[7];
  const float* gamma = (const float*)d_in[8];
  const float* beta  = (const float*)d_in[9];
  float* out = (float*)d_out;

  char* ws = (char*)d_ws;
  float* scores = (float*)(ws);                       // 2 MB  (then att in place)
  float* agg    = (float*)(ws + (2u << 20));          // 2 MB
  float* pre    = (float*)(ws + (4u << 20));          // 2 MB
  float* stats  = (float*)(ws + (6u << 20));          // 512 f32

  const size_t smem1 = 1024 + (size_t)D256 * WT_STRIDE * 2;   // 136,192 B
  const size_t smem3 = (size_t)D256 * WT_STRIDE * 2;          // 135,168 B
  const size_t smem4 = 2u * D256 * WT_STRIDE * 2;             // 270,336 B
  (void)hipFuncSetAttribute((const void*)k_scores, hipFuncAttributeMaxDynamicSharedMemorySize, (int)smem1);
  (void)hipFuncSetAttribute((const void*)k_agg,    hipFuncAttributeMaxDynamicSharedMemorySize, (int)smem3);
  (void)hipFuncSetAttribute((const void*)k_out,    hipFuncAttributeMaxDynamicSharedMemorySize, (int)smem4);

  k_scores <<<2048, 256, smem1, stream>>>(x, Wp, bp, watt, scores);
  k_softmax<<<32,   256, 0,     stream>>>(scores);
  k_agg    <<<16,   256, smem3, stream>>>(scores, x, agg);
  k_out    <<<16,   256, smem4, stream>>>(agg, x, W1, b1, W2, b2, pre);
  k_stats  <<<256,  256, 0,     stream>>>(pre, stats);
  k_bn_selu<<<2048, 256, 0,     stream>>>(pre, stats, gamma, beta, out);
}
